// Flow_6081673691683
// MI455X (gfx1250) — matrix operand order repaired and checked by reading, not compiled, not run on hardware
//
#include <hip/hip_runtime.h>
#include <math.h>

#define A_DIMV  12
#define AUG     24
#define OBS_D   48
#define HID     256
#define NSTEPS  10
#define PBLEND  0.95f
#define QBLEND  0.05f
#define DTSTEP  0.1f
#define BATCH   2048
#define TILE    16
#define THREADS 256

typedef __attribute__((ext_vector_type(16))) __bf16 v16bf;
typedef __attribute__((ext_vector_type(8)))  __bf16 v8bf;
typedef __attribute__((ext_vector_type(8)))  float  v8f;

// ---------------- workspace (bf16) element offsets ----------------
#define WS_W1T 0                         // [256][64]  W1t[n][k] = W1[k][n], k>=61 -> 0
#define WS_W2T (WS_W1T + 256*64)         // [256][256] W2t[n][k] = W2[k][n]
#define WS_W3T (WS_W2T + 256*256)        // [256][256]
#define WS_W4T (WS_W3T + 256*256)        // [16][256]  W4t[n][k] = W4[k][n], n>=12 -> 0
#define WS_W1X (WS_W4T + 16*256)         // [16][256]  W1x[i][j] = W1[i][j], i>=12 -> 0
#define WS_TOTAL (WS_W1X + 16*256)       // 155648 elements

// ---------------- dynamic LDS byte offsets ----------------
enum {
  SM_IN  = 0,                       // bf16 [16][64]
  SM_H1  = SM_IN  + TILE*64*2,      // bf16 [16][256]
  SM_H2  = SM_H1  + TILE*HID*2,
  SM_H3  = SM_H2  + TILE*HID*2,
  SM_TA  = SM_H3  + TILE*HID*2,     // bf16 [16][256] tangent ping
  SM_TB  = SM_TA  + TILE*HID*2,     // bf16 [16][256] tangent pong
  SM_ML  = SM_TB  + TILE*HID*2,     // f32  [16][12][12]
  SM_G   = SM_ML  + TILE*144*4,     // f32  [16][24][24]
  SM_ZS  = SM_G   + TILE*576*4,     // f32  [16][12] x6 state buffers
  SM_YS  = SM_ZS  + TILE*A_DIMV*4,
  SM_ZI  = SM_YS  + TILE*A_DIMV*4,
  SM_YI  = SM_ZI  + TILE*A_DIMV*4,
  SM_ZT  = SM_YI  + TILE*A_DIMV*4,
  SM_YT  = SM_ZT  + TILE*A_DIMV*4,
  SM_TOTAL = SM_YT + TILE*A_DIMV*4  // ~93.7 KB
};

// ---------------- helpers ----------------
__device__ __forceinline__ v8f zero8() {
  v8f c;
#pragma unroll
  for (int i = 0; i < 8; ++i) c[i] = 0.f;
  return c;
}

__device__ __forceinline__ v8f wmma_bf16(v16bf a, v16bf b, v8f c) {
  return __builtin_amdgcn_wmma_f32_16x16x32_bf16(false, a, false, b, (short)0, c,
                                                 false, false);
}

// branch-free tanh: copysign((1-e)/(1+e), x), e = 2^(-2*log2(e)*|x|)
// one v_exp_f32 + one v_rcp_f32, no exec-mask divergence; bf16-accurate.
__device__ __forceinline__ float fast_tanh(float x) {
  float e = __builtin_amdgcn_exp2f(-2.8853900817779268f * fabsf(x));
  float r = (1.f - e) * __builtin_amdgcn_rcpf(1.f + e);
  return copysignf(r, x);
}

// A fragment: 16x32 bf16 tile from row-major [rows][ld]; lane = M, ISA K layout.
__device__ __forceinline__ v16bf load_a_tile(const __bf16* base, int row0, int ld, int k0) {
  int lane = threadIdx.x & 31;
  int m = lane & 15, hh = lane >> 4;
  const __bf16* p = base + (size_t)(row0 + m) * ld + k0 + hh * 8;
  v8bf lo = *(const v8bf*)p;          // K = k0 + hh*8 + [0,8)
  v8bf hi = *(const v8bf*)(p + 16);   // K = k0 + 16 + hh*8 + [0,8)
  v16bf a;
#pragma unroll
  for (int e = 0; e < 8; ++e) { a[e] = lo[e]; a[e + 8] = hi[e]; }
  return a;
}

// B fragment: 32x16 bf16 tile from K-contiguous transposed weights Wt[n][k]; lane = N.
__device__ __forceinline__ v16bf load_b_tile(const __bf16* baseT, int n0, int ld, int k0) {
  int lane = threadIdx.x & 31;
  int n = lane & 15, hh = lane >> 4;
  const __bf16* p = baseT + (size_t)(n0 + n) * ld + k0 + hh * 8;
  v8bf lo = *(const v8bf*)p;          // K = k0 + hh*8 + [0,8)
  v8bf hi = *(const v8bf*)(p + 16);   // K = k0 + 16 + hh*8 + [0,8)
  v16bf b;
#pragma unroll
  for (int e = 0; e < 8; ++e) { b[e] = lo[e]; b[e + 8] = hi[e]; }
  return b;
}

// Tangent layer-1 seed: A[i][k] = W1x[i][k] * (1 - h1[k]^2)
__device__ __forceinline__ v16bf load_a_tan1(const __bf16* W1X, const __bf16* h1s, int k0) {
  int lane = threadIdx.x & 31;
  int m = lane & 15, hh = lane >> 4;
  const __bf16* p = W1X + (size_t)m * HID;
  v16bf a;
#pragma unroll
  for (int e = 0; e < 16; ++e) {
    int k = k0 + ((e < 8) ? (hh * 8 + e) : (16 + hh * 8 + (e - 8)));
    float hv = (float)h1s[k];
    a[e] = (__bf16)((float)p[k] * (1.f - hv * hv));
  }
  return a;
}

// ---------------- layer routines (A fragments hoisted across both N-tiles) --------
template <int K>
__device__ void fwd_hidden_layer(const __bf16* act_in, const __bf16* WT,
                                 const float* bias, __bf16* h_out) {
  int wave = threadIdx.x >> 5;
  int lane = threadIdx.x & 31;
  int n_lane = lane & 15, hh = lane >> 4;
  v16bf afrag[K >> 5];
#pragma unroll
  for (int kt = 0; kt < (K >> 5); ++kt) afrag[kt] = load_a_tile(act_in, 0, K, kt * 32);
#pragma unroll 1
  for (int t = 0; t < 2; ++t) {
    int nt = wave * 2 + t;
    v8f c = zero8();
#pragma unroll
    for (int kt = 0; kt < (K >> 5); ++kt) {
      v16bf b = load_b_tile(WT, nt * 16, K, kt * 32);
      c = wmma_bf16(afrag[kt], b, c);
    }
    int col = nt * 16 + n_lane;
    float bv = bias[col];
#pragma unroll
    for (int v = 0; v < 8; ++v) {
      int row = v + 8 * hh;
      h_out[row * HID + col] = (__bf16)fast_tanh(c[v] + bv);
    }
  }
}

__device__ void fwd_final_layer(const __bf16* h3, const __bf16* W4T,
                                const float* b4, float* outv) {
  int lane = threadIdx.x & 31;
  int n = lane & 15, hh = lane >> 4;
  v8f c = zero8();
#pragma unroll
  for (int kt = 0; kt < 8; ++kt) {
    v16bf a = load_a_tile(h3, 0, HID, kt * 32);
    v16bf b = load_b_tile(W4T, 0, HID, kt * 32);
    c = wmma_bf16(a, b, c);
  }
  if (n < A_DIMV) {
    float bv = b4[n];
#pragma unroll
    for (int v = 0; v < 8; ++v) {
      int s = v + 8 * hh;                 // sample row
      outv[s * A_DIMV + n] = c[v] + bv;
    }
  }
}

__device__ void tan_layer2(const __bf16* W1X, const __bf16* h1s, const __bf16* W2T,
                           const __bf16* h2s, __bf16* Tout) {
  int wave = threadIdx.x >> 5;
  int lane = threadIdx.x & 31;
  int n_lane = lane & 15, hh = lane >> 4;
  v16bf afrag[8];
#pragma unroll
  for (int kt = 0; kt < 8; ++kt) afrag[kt] = load_a_tan1(W1X, h1s, kt * 32);
#pragma unroll 1
  for (int t = 0; t < 2; ++t) {
    int nt = wave * 2 + t;
    v8f c = zero8();
#pragma unroll
    for (int kt = 0; kt < 8; ++kt) {
      v16bf b = load_b_tile(W2T, nt * 16, HID, kt * 32);
      c = wmma_bf16(afrag[kt], b, c);
    }
    int col = nt * 16 + n_lane;
    float hv = (float)h2s[col];
    float d = 1.f - hv * hv;
#pragma unroll
    for (int v = 0; v < 8; ++v) {
      int row = v + 8 * hh;
      Tout[row * HID + col] = (__bf16)(c[v] * d);
    }
  }
}

__device__ void tan_layer3(const __bf16* Tin, const __bf16* W3T,
                           const __bf16* h3s, __bf16* Tout) {
  int wave = threadIdx.x >> 5;
  int lane = threadIdx.x & 31;
  int n_lane = lane & 15, hh = lane >> 4;
  v16bf afrag[8];
#pragma unroll
  for (int kt = 0; kt < 8; ++kt) afrag[kt] = load_a_tile(Tin, 0, HID, kt * 32);
#pragma unroll 1
  for (int t = 0; t < 2; ++t) {
    int nt = wave * 2 + t;
    v8f c = zero8();
#pragma unroll
    for (int kt = 0; kt < 8; ++kt) {
      v16bf b = load_b_tile(W3T, nt * 16, HID, kt * 32);
      c = wmma_bf16(afrag[kt], b, c);
    }
    int col = nt * 16 + n_lane;
    float hv = (float)h3s[col];
    float d = 1.f - hv * hv;
#pragma unroll
    for (int v = 0; v < 8; ++v) {
      int row = v + 8 * hh;
      Tout[row * HID + col] = (__bf16)(c[v] * d);
    }
  }
}

__device__ void tan_final(int s, const __bf16* Tin, const __bf16* W4T, float* Ml) {
  int lane = threadIdx.x & 31;
  int n = lane & 15, hh = lane >> 4;
  v8f c = zero8();
#pragma unroll
  for (int kt = 0; kt < 8; ++kt) {
    v16bf a = load_a_tile(Tin, 0, HID, kt * 32);
    v16bf b = load_b_tile(W4T, 0, HID, kt * 32);
    c = wmma_bf16(a, b, c);
  }
  if (n < A_DIMV) {
#pragma unroll
    for (int v = 0; v < 8; ++v) {
      int i = v + 8 * hh;                 // tangent direction (input dim)
      if (i < A_DIMV) Ml[s * 144 + i * 12 + n] = c[v];  // M[i][o] = d out_o / d x_i
    }
  }
}

// ---------------- shared-memory view ----------------
struct Shm {
  __bf16 *inb, *h1, *h2, *h3, *tA, *tB;
  float *Ml, *G, *zs, *ys, *zin, *yin, *zt, *yt;
};

// One full MLP eval (forward + per-sample input-Jacobians M[s])
__device__ void eval_mlp(const Shm& S, const float* xsrc, float tval,
                         const float* obs, int sb,
                         const __bf16* W1T, const __bf16* W2T, const __bf16* W3T,
                         const __bf16* W4T, const __bf16* W1X,
                         const float* b1, const float* b2, const float* b3,
                         const float* b4, float* outv) {
  int tid = threadIdx.x;
  // build padded input rows [x(12) | obs(48) | t | 0,0,0] as bf16
  for (int e = tid; e < TILE * 64; e += THREADS) {
    int s = e >> 6, k = e & 63;
    float v;
    if (k < A_DIMV)      v = xsrc[s * A_DIMV + k];
    else if (k < 60)     v = obs[(size_t)(sb + s) * OBS_D + (k - A_DIMV)];
    else if (k == 60)    v = tval;
    else                 v = 0.f;
    S.inb[e] = (__bf16)v;
  }
  __syncthreads();
  fwd_hidden_layer<64 >(S.inb, W1T, b1, S.h1); __syncthreads();
  fwd_hidden_layer<HID>(S.h1,  W2T, b2, S.h2); __syncthreads();
  fwd_hidden_layer<HID>(S.h2,  W3T, b3, S.h3); __syncthreads();
  if ((tid >> 5) == 0) fwd_final_layer(S.h3, W4T, b4, outv);
  __syncthreads();
  // tangent chain, one sample at a time (all 8 waves cooperate on L2/L3)
  for (int s = 0; s < TILE; ++s) {
    tan_layer2(W1X, S.h1 + s * HID, W2T, S.h2 + s * HID, S.tA);
    __syncthreads();
    tan_layer3(S.tA, W3T, S.h3 + s * HID, S.tB);
    __syncthreads();
    if ((tid >> 5) == (s & 7)) tan_final(s, S.tB, W4T, S.Ml);
  }
  __syncthreads();
}

// ---------------- weight conversion (f32 -> bf16, transposed/padded) ----------------
__global__ void conv_weights(const float* __restrict__ W1, const float* __restrict__ W2,
                             const float* __restrict__ W3, const float* __restrict__ W4,
                             __bf16* __restrict__ ws) {
  int idx = blockIdx.x * blockDim.x + threadIdx.x;
  if (idx >= WS_TOTAL) return;
  if (idx < WS_W2T) {                                // W1T [256][64]
    int r = idx - WS_W1T, n = r >> 6, k = r & 63;
    ws[idx] = (__bf16)((k < 61) ? W1[k * 256 + n] : 0.f);
  } else if (idx < WS_W3T) {                         // W2T [256][256]
    int r = idx - WS_W2T, n = r >> 8, k = r & 255;
    ws[idx] = (__bf16)W2[k * 256 + n];
  } else if (idx < WS_W4T) {                         // W3T [256][256]
    int r = idx - WS_W3T, n = r >> 8, k = r & 255;
    ws[idx] = (__bf16)W3[k * 256 + n];
  } else if (idx < WS_W1X) {                         // W4T [16][256]
    int r = idx - WS_W4T, n = r >> 8, k = r & 255;
    ws[idx] = (__bf16)((n < A_DIMV) ? W4[k * A_DIMV + n] : 0.f);
  } else {                                           // W1X [16][256]
    int r = idx - WS_W1X, i = r >> 8, j = r & 255;
    ws[idx] = (__bf16)((i < A_DIMV) ? W1[i * 256 + j] : 0.f);
  }
}

// ---------------- main kernel ----------------
__global__ __launch_bounds__(THREADS) void flow_heun_kernel(
    const float* __restrict__ obs, const float* __restrict__ x0,
    const float* __restrict__ b1, const float* __restrict__ b2,
    const float* __restrict__ b3, const float* __restrict__ b4,
    const __bf16* __restrict__ ws, float* __restrict__ out) {
  extern __shared__ __align__(16) char smem[];
  Shm S;
  S.inb = (__bf16*)(smem + SM_IN);
  S.h1  = (__bf16*)(smem + SM_H1);
  S.h2  = (__bf16*)(smem + SM_H2);
  S.h3  = (__bf16*)(smem + SM_H3);
  S.tA  = (__bf16*)(smem + SM_TA);
  S.tB  = (__bf16*)(smem + SM_TB);
  S.Ml  = (float*)(smem + SM_ML);
  S.G   = (float*)(smem + SM_G);
  S.zs  = (float*)(smem + SM_ZS);
  S.ys  = (float*)(smem + SM_YS);
  S.zin = (float*)(smem + SM_ZI);
  S.yin = (float*)(smem + SM_YI);
  S.zt  = (float*)(smem + SM_ZT);
  S.yt  = (float*)(smem + SM_YT);

  const __bf16* W1T = ws + WS_W1T;
  const __bf16* W2T = ws + WS_W2T;
  const __bf16* W3T = ws + WS_W3T;
  const __bf16* W4T = ws + WS_W4T;
  const __bf16* W1X = ws + WS_W1X;

  int tid = threadIdx.x;
  int sb  = blockIdx.x * TILE;

  // warm L2 for the bf16 weights
  __builtin_prefetch(W2T + (size_t)tid * 256, 0, 1);
  __builtin_prefetch(W3T + (size_t)tid * 256, 0, 1);

  // init state & G = I24 per sample
  for (int e = tid; e < TILE * 576; e += THREADS) {
    int r = (e % 576) / 24, c = e % 24;
    S.G[e] = (r == c) ? 1.f : 0.f;
  }
  for (int e = tid; e < TILE * A_DIMV; e += THREADS) {
    int s = e / A_DIMV, j = e % A_DIMV;
    S.zs[e] = x0[(size_t)(sb + s) * AUG + j];
    S.ys[e] = x0[(size_t)(sb + s) * AUG + A_DIMV + j];
  }
  __syncthreads();

  for (int step = 0; step < NSTEPS; ++step) {
    float tval = (float)step / (float)NSTEPS;

    // ---- eval 1: z_t = mlp(y); M1 ----
    eval_mlp(S, S.ys, tval, obs, sb, W1T, W2T, W3T, W4T, W1X, b1, b2, b3, b4, S.zt);
    // Gz += dt * M1^T Gy ; z_in = z + dt*z_t   (reads rows 12..23, writes rows 0..11)
    for (int e = tid; e < TILE * 288; e += THREADS) {
      int s = e / 288, r = e % 288, o = r / 24, cc = r % 24;
      float acc = 0.f;
#pragma unroll
      for (int i = 0; i < A_DIMV; ++i)
        acc += S.Ml[s * 144 + i * 12 + o] * S.G[s * 576 + (12 + i) * 24 + cc];
      S.G[s * 576 + o * 24 + cc] += DTSTEP * acc;
    }
    for (int e = tid; e < TILE * A_DIMV; e += THREADS)
      S.zin[e] = S.zs[e] + DTSTEP * S.zt[e];
    __syncthreads();

    // ---- eval 2: y_t = mlp(z_in); M2 ----
    eval_mlp(S, S.zin, tval, obs, sb, W1T, W2T, W3T, W4T, W1X, b1, b2, b3, b4, S.yt);
    // Gy += dt * M2^T Gz_in ; y_in = y + dt*y_t
    for (int e = tid; e < TILE * 288; e += THREADS) {
      int s = e / 288, r = e % 288, o = r / 24, cc = r % 24;
      float acc = 0.f;
#pragma unroll
      for (int i = 0; i < A_DIMV; ++i)
        acc += S.Ml[s * 144 + i * 12 + o] * S.G[s * 576 + i * 24 + cc];
      S.G[s * 576 + (12 + o) * 24 + cc] += DTSTEP * acc;
    }
    for (int e = tid; e < TILE * A_DIMV; e += THREADS)
      S.yin[e] = S.ys[e] + DTSTEP * S.yt[e];
    __syncthreads();

    // ---- Heun blend of state and Jacobian ----
    for (int e = tid; e < TILE * 288; e += THREADS) {
      int s = e / 288, r = e % 288, o = r / 24, cc = r % 24;
      float gz = S.G[s * 576 + o * 24 + cc];
      float gy = S.G[s * 576 + (12 + o) * 24 + cc];
      float gzn = PBLEND * gz + QBLEND * gy;
      float gyn = PBLEND * gy + QBLEND * gzn;
      S.G[s * 576 + o * 24 + cc] = gzn;
      S.G[s * 576 + (12 + o) * 24 + cc] = gyn;
    }
    for (int e = tid; e < TILE * A_DIMV; e += THREADS) {
      float zi = S.zin[e], yi = S.yin[e];
      float zn = PBLEND * zi + QBLEND * yi;
      float yn = PBLEND * yi + QBLEND * zn;
      S.zs[e] = zn;
      S.ys[e] = yn;
    }
    __syncthreads();
  }

  // ---- outputs: action_aug ----
  for (int e = tid; e < TILE * AUG; e += THREADS) {
    int s = e / AUG, j = e % AUG;
    out[(size_t)(sb + s) * AUG + j] =
        (j < A_DIMV) ? S.zs[s * A_DIMV + j] : S.ys[s * A_DIMV + (j - A_DIMV)];
  }
  __syncthreads();

  // ---- log_probs: -0.5*|x0|^2 - 12*log(2pi) - logabsdet(G) via in-place LU ----
  if (tid < TILE) {
    float* Am = S.G + tid * 576;   // 24x24
    float ld = 0.f;
    for (int k = 0; k < 24; ++k) {
      int piv = k;
      float mx = fabsf(Am[k * 24 + k]);
      for (int r = k + 1; r < 24; ++r) {
        float v = fabsf(Am[r * 24 + k]);
        if (v > mx) { mx = v; piv = r; }
      }
      if (piv != k) {
        for (int c = k; c < 24; ++c) {
          float tmp = Am[k * 24 + c];
          Am[k * 24 + c] = Am[piv * 24 + c];
          Am[piv * 24 + c] = tmp;
        }
      }
      float d = Am[k * 24 + k];
      ld += logf(fabsf(d));
      float inv = 1.f / d;
      for (int r = k + 1; r < 24; ++r) {
        float f = Am[r * 24 + k] * inv;
        for (int c = k + 1; c < 24; ++c) Am[r * 24 + c] -= f * Am[k * 24 + c];
      }
    }
    float ss = 0.f;
    const float* xp = x0 + (size_t)(sb + tid) * AUG;
    for (int j = 0; j < AUG; ++j) { float v = xp[j]; ss += v * v; }
    float lp = -0.5f * ss - 12.0f * logf(6.283185307179586f) - ld;
    out[(size_t)BATCH * AUG + (sb + tid)] = lp;
  }
}

// ---------------- host entry ----------------
extern "C" void kernel_launch(void* const* d_in, const int* in_sizes, int n_in,
                              void* d_out, int out_size, void* d_ws, size_t ws_size,
                              hipStream_t stream) {
  const float* obs = (const float*)d_in[0];
  const float* x0  = (const float*)d_in[1];
  const float* W1  = (const float*)d_in[2];
  const float* b1  = (const float*)d_in[3];
  const float* W2  = (const float*)d_in[4];
  const float* b2  = (const float*)d_in[5];
  const float* W3  = (const float*)d_in[6];
  const float* b3  = (const float*)d_in[7];
  const float* W4  = (const float*)d_in[8];
  const float* b4  = (const float*)d_in[9];
  __bf16* ws = (__bf16*)d_ws;
  float* out = (float*)d_out;

  int convBlocks = (WS_TOTAL + 255) / 256;
  conv_weights<<<convBlocks, 256, 0, stream>>>(W1, W2, W3, W4, ws);

  flow_heun_kernel<<<BATCH / TILE, THREADS, SM_TOTAL, stream>>>(
      obs, x0, b1, b2, b3, b4, ws, out);
}
